// ConvX_90598040141881
// MI455X (gfx1250) — matrix operand order repaired and checked by reading, not compiled, not run on hardware
//
#include <hip/hip_runtime.h>
#include <hip/hip_bf16.h>
#include <cstdint>

// Shapes from reference
#define T_ 16
#define C_ 64
#define H_ 384
#define W_ 768
#define F_ 5
#define KTOT (C_ * F_ * F_)   // 1600
#define NKB (KTOT / 32)       // 50 k-blocks of 32
#define BPX 64                // pixels along W per workgroup
#define WTILES (W_ / BPX)     // 12
#define HALO (BPX + F_ - 1)   // 68 columns staged in LDS per row
#define PIXB 144              // padded bytes per pixel slot (128 data + 16 pad)
#define NPIX (F_ * HALO)      // 340 staged pixels
#define SMEM_BYTES (NPIX * PIXB)   // 48960 B

typedef __attribute__((ext_vector_type(16))) _Float16 v16h;
typedef __attribute__((ext_vector_type(8)))  _Float16 v8h;
typedef __attribute__((ext_vector_type(8)))  float    v8f;
typedef __attribute__((address_space(3))) char* lds_ptr_t;

// ---------------------------------------------------------------------------
// Kernel 1: pack w (Cout,Cin,5,5) f32 into WMMA A-fragment layout f16.
// Fragment (ct, kb): 32 lanes x 16 f16 contiguous (1024 B). ISA 16-bit A 16x32:
// lane<16 -> M=lane, K={0..7,16..23}; lane>=16 -> M=lane-16, K={8..15,24..31};
// element e -> relK = (e<8 ? e : e+8) + (lane>=16)*8.
// Global K order: K = tap*64 + ci, tap = fy*5+fx  (kb = tap*2 + cb).
// ---------------------------------------------------------------------------
__global__ __launch_bounds__(256) void pack_weights(const float* __restrict__ w,
                                                    _Float16* __restrict__ wA) {
  int g = blockIdx.x * blockDim.x + threadIdx.x;
  if (g >= 4 * NKB * 32) return;
  int lane = g & 31;
  int frag = g >> 5;                 // ct*NKB + kb
  int ct = frag / NKB;
  int kb = frag % NKB;
  int hi = lane >> 4;
  int co = ct * 16 + (lane & 15);
  _Float16* dst = wA + (size_t)frag * 512 + lane * 16;
#pragma unroll
  for (int e = 0; e < 16; ++e) {
    int relK = (e < 8 ? e : e + 8) + hi * 8;
    int K    = kb * 32 + relK;
    int ci   = K & 63;
    int tap  = K >> 6;
    int fy   = tap / 5, fx = tap % 5;
    dst[e] = (_Float16)w[co * KTOT + ci * 25 + fy * 5 + fx];
  }
}

// ---------------------------------------------------------------------------
// Kernel 2: x NCHW f32 -> xT NHWC f16 via LDS tile (coalesced read & write).
// ---------------------------------------------------------------------------
__global__ __launch_bounds__(256) void transpose_nchw_to_nhwc(
    const float* __restrict__ x, _Float16* __restrict__ xT) {
  __shared__ _Float16 tile[64 * 65];
  int bid = blockIdx.x;
  int wb = bid % (W_ / 64); bid /= (W_ / 64);
  int h  = bid % H_;        bid /= H_;
  int t  = bid;
  int w0 = wb * 64;
  int tid = threadIdx.x;
#pragma unroll
  for (int i = 0; i < 16; ++i) {       // lanes sweep w: coalesced f32 reads
    int e = i * 256 + tid;
    int c = e >> 6, w = e & 63;
    tile[c * 65 + w] =
        (_Float16)x[(((size_t)t * C_ + c) * H_ + h) * W_ + w0 + w];
  }
  __syncthreads();
#pragma unroll
  for (int i = 0; i < 16; ++i) {       // lanes sweep c: coalesced f16 writes
    int e = i * 256 + tid;
    int w = e >> 6, c = e & 63;
    xT[(((size_t)t * H_ + h) * W_ + w0 + w) * C_ + c] = tile[c * 65 + w];
  }
}

// ---------------------------------------------------------------------------
// Kernel 3: implicit-GEMM conv. Block = 256 threads = 8 waves = 4 co-tiles x
// 2 pixel-groups; each wave: 16co x 32px (two 16x16 WMMA tiles sharing A).
// Halo (5 rows x 68 cols x 64ci, wrap folded in) staged in LDS via the
// gfx1250 async copy path (global_load_async_to_lds_b128 + s_wait_asynccnt).
// ---------------------------------------------------------------------------
__device__ inline v16h lds_frag(const char* smem, int fy, int col, int cioff) {
  const char* p = smem + (fy * HALO + col) * PIXB + cioff;
  v8h lo = *(const v8h*)(p);
  v8h hi = *(const v8h*)(p + 32);
  return __builtin_shufflevector(lo, hi, 0, 1, 2, 3, 4, 5, 6, 7,
                                 8, 9, 10, 11, 12, 13, 14, 15);
}

__global__ __launch_bounds__(256) void conv_wmma(
    const _Float16* __restrict__ xT, const _Float16* __restrict__ wA,
    const float* __restrict__ bias, float* __restrict__ y) {
  __shared__ uint4 smem4[SMEM_BYTES / 16];
  char* smem = (char*)smem4;

  const int tid  = threadIdx.x;
  const int wave = tid >> 5;
  const int lane = tid & 31;
  const int ct = wave & 3;             // co tile 0..3
  const int pg = wave >> 2;            // pixel group 0..1 (32 px each)

  int flat = blockIdx.x;
  const int wt = flat % WTILES; flat /= WTILES;
  const int h  = flat % H_;     flat /= H_;
  const int t  = flat;
  const int w0 = wt * BPX;

  // ---- cooperative halo fill: async global->LDS, 16B per lane per issue.
  // ASYNCcnt tracks completion.
  for (int idx = tid; idx < NPIX * 8; idx += 256) {
    int j = idx & 7;                   // 16B chunk within 128B pixel payload
    int p = idx >> 3;                  // staged pixel 0..339
    int row = p / HALO;
    int col = p % HALO;
    int hh = h + row - 2;
    hh += (hh < 0) ? H_ : 0;
    hh -= (hh >= H_) ? H_ : 0;
    int ww = w0 + col - 2;
    ww += (ww < 0) ? W_ : 0;
    ww -= (ww >= W_) ? W_ : 0;
    uint64_t gaddr = (uint64_t)((const char*)xT +
        ((((size_t)t * H_ + hh) * W_ + ww) * C_) * 2 + j * 16);
    lds_ptr_t dsaddr = (lds_ptr_t)(smem + p * PIXB + j * 16);
    asm volatile("global_load_async_to_lds_b128 %0, %1, off"
                 :: "v"(dsaddr), "v"(gaddr) : "memory");
  }
  asm volatile("s_wait_asynccnt 0x0" ::: "memory");
  __syncthreads();

  const int laneN = lane & 15;
  const int hi    = lane >> 4;         // 0: K/M low half ; 1: K/M high half
  const int colBase = pg * 32 + laneN; // local column of first tile

  v8f acc0 = {};
  v8f acc1 = {};
  const _Float16* aPtr = wA + (size_t)(ct * NKB) * 512 + lane * 16;

#pragma unroll
  for (int fy = 0; fy < F_; ++fy) {
#pragma unroll
    for (int fx = 0; fx < F_; ++fx) {
#pragma unroll
      for (int cb = 0; cb < 2; ++cb) {               // kb = (fy*5+fx)*2 + cb
        __builtin_prefetch((const void*)(aPtr + 2048), 0, 1);
        v16h a = *(const v16h*)(aPtr); aPtr += 512;  // 1 KB fragment stride
        int cio = cb * 64 + hi * 16;                 // ci byte offset for lane
        v16h b0 = lds_frag(smem, fy, colBase + fx, cio);
        v16h b1 = lds_frag(smem, fy, colBase + 16 + fx, cio);
        acc0 = __builtin_amdgcn_wmma_f32_16x16x32_f16(false, a, false, b0,
                                                      (short)0, acc0, false, false);
        acc1 = __builtin_amdgcn_wmma_f32_16x16x32_f16(false, a, false, b1,
                                                      (short)0, acc1, false, false);
      }
    }
  }

  // C/D layout: VGPR j, lanes 0-15 -> M=j ; lanes 16-31 -> M=j+8 ; N=lane%16
  const int coBase = ct * 16 + hi * 8;
  float* y0 = y + (((size_t)t * C_ + coBase) * H_ + h) * W_ + w0 + pg * 32 + laneN;
#pragma unroll
  for (int j = 0; j < 8; ++j) {
    float bj = bias[coBase + j];
    y0[(size_t)j * H_ * W_]      = acc0[j] + bj;
    y0[(size_t)j * H_ * W_ + 16] = acc1[j] + bj;
  }
}

// ---------------------------------------------------------------------------
extern "C" void kernel_launch(void* const* d_in, const int* in_sizes, int n_in,
                              void* d_out, int out_size, void* d_ws, size_t ws_size,
                              hipStream_t stream) {
  const float* x = (const float*)d_in[0];   // (16,64,384,768) f32
  const float* w = (const float*)d_in[1];   // (64,64,5,5)     f32
  const float* b = (const float*)d_in[2];   // (64,)           f32
  float* y = (float*)d_out;                 // (16,64,384,768) f32

  // workspace: [packed weights f16: 204800 B][xT NHWC f16: 604 MB]
  _Float16* wA = (_Float16*)d_ws;
  _Float16* xT = (_Float16*)((char*)d_ws + (256 * 1024));

  pack_weights<<<(4 * NKB * 32 + 255) / 256, 256, 0, stream>>>(w, wA);
  transpose_nchw_to_nhwc<<<T_ * H_ * (W_ / 64), 256, 0, stream>>>(x, xT);
  conv_wmma<<<T_ * H_ * WTILES, 256, 0, stream>>>(xT, wA, b, y);
}
